// SelfAttention_89034672046916
// MI455X (gfx1250) — 16-bit and 8-bit matrix operand order repaired and checked by reading, compiled, not run on hardware
//
#include <hip/hip_runtime.h>
#include <hip/hip_bf16.h>

typedef __attribute__((ext_vector_type(16))) _Float16 v16h;
typedef __attribute__((ext_vector_type(8)))  _Float16 h8;
typedef __attribute__((ext_vector_type(8)))  float    v8f;
typedef __attribute__((ext_vector_type(8)))  int      v8i;

#define DIM   512
#define SEQ   2048
#define NBAT  8
#define NTOK  16384
#define SCORE_SCALE 0.35355339059327373f

static __device__ __forceinline__ v8i mk8i(uint4 a, uint4 b) {
  v8i r;
  r[0]=(int)a.x; r[1]=(int)a.y; r[2]=(int)a.z; r[3]=(int)a.w;
  r[4]=(int)b.x; r[5]=(int)b.y; r[6]=(int)b.z; r[7]=(int)b.w;
  return r;
}
static __device__ __forceinline__ v8i mk8i2(uint2 a, uint2 b, uint2 c, uint2 d) {
  v8i r;
  r[0]=(int)a.x; r[1]=(int)a.y; r[2]=(int)b.x; r[3]=(int)b.y;
  r[4]=(int)c.x; r[5]=(int)c.y; r[6]=(int)d.x; r[7]=(int)d.y;
  return r;
}
static __device__ __forceinline__ v16h mk16h(h8 lo, h8 hi) {
  v16h r;
  #pragma unroll
  for (int i = 0; i < 8; ++i) { r[i] = lo[i]; r[8 + i] = hi[i]; }
  return r;
}

__global__ __launch_bounds__(128) void k_actquant(const float* __restrict__ x,
                                                  signed char* __restrict__ xq,
                                                  float* __restrict__ ascale) {
  const int t   = blockIdx.x;
  const int tid = threadIdx.x;
  const float4 v = reinterpret_cast<const float4*>(x + (size_t)t * DIM)[tid];
  float m = fmaxf(fmaxf(fabsf(v.x), fabsf(v.y)), fmaxf(fabsf(v.z), fabsf(v.w)));
  #pragma unroll
  for (int off = 16; off > 0; off >>= 1) m = fmaxf(m, __shfl_xor(m, off, 32));
  __shared__ float sm[4];
  if ((tid & 31) == 0) sm[tid >> 5] = m;
  __syncthreads();
  m = fmaxf(fmaxf(sm[0], sm[1]), fmaxf(sm[2], sm[3]));
  m = fmaxf(m, 1e-5f);
  const float s = 127.0f / m;
  int q0 = (int)rintf(v.x * s), q1 = (int)rintf(v.y * s);
  int q2 = (int)rintf(v.z * s), q3 = (int)rintf(v.w * s);
  q0 = min(127, max(-128, q0)); q1 = min(127, max(-128, q1));
  q2 = min(127, max(-128, q2)); q3 = min(127, max(-128, q3));
  const unsigned pack = (unsigned)(q0 & 255) | ((unsigned)(q1 & 255) << 8) |
                        ((unsigned)(q2 & 255) << 16) | ((unsigned)(q3 & 255) << 24);
  reinterpret_cast<unsigned*>(xq + (size_t)t * DIM)[tid] = pack;
  if (tid == 0) ascale[t] = m * (1.0f / 127.0f);
}

__global__ __launch_bounds__(256) void k_wquant(const float* __restrict__ Wq,
                                                const float* __restrict__ Wk,
                                                const float* __restrict__ Wv,
                                                signed char* __restrict__ wq3,
                                                float* __restrict__ wsinv) {
  const float* W = (blockIdx.x == 0) ? Wq : (blockIdx.x == 1) ? Wk : Wv;
  signed char* o = wq3 + (size_t)blockIdx.x * DIM * DIM;
  const int tid = threadIdx.x;
  float acc = 0.0f;
  for (int i = tid; i < DIM * DIM; i += 256) acc += fabsf(W[i]);
  #pragma unroll
  for (int off = 16; off > 0; off >>= 1) acc += __shfl_xor(acc, off, 32);
  __shared__ float sm[8];
  if ((tid & 31) == 0) sm[tid >> 5] = acc;
  __syncthreads();
  float tot = 0.0f;
  #pragma unroll
  for (int i = 0; i < 8; ++i) tot += sm[i];
  const float mean = fmaxf(tot * (1.0f / (float)(DIM * DIM)), 1e-5f);
  const float s = 1.0f / mean;
  if (tid == 0) wsinv[blockIdx.x] = mean;
  for (int i = tid; i < DIM * DIM; i += 256) {
    int qv = (int)rintf(W[i] * s);
    qv = qv > 1 ? 1 : (qv < -1 ? -1 : qv);
    o[i] = (signed char)qv;
  }
}

__global__ __launch_bounds__(256) void k_proj(const signed char* __restrict__ xq,
                                              const float* __restrict__ ascale,
                                              const signed char* __restrict__ wq3,
                                              const float* __restrict__ wsinv,
                                              _Float16* __restrict__ qf,
                                              _Float16* __restrict__ kf,
                                              _Float16* __restrict__ vT) {
  const int lane = threadIdx.x & 31;
  const int w    = threadIdx.x >> 5;
  const int ln   = lane & 15;
  const int lh   = lane >> 4;
  const int tt   = blockIdx.x;
  const int ob   = blockIdx.y * 128 + w * 16;

  const signed char* wqp = wq3;
  const signed char* wkp = wq3 + (size_t)DIM * DIM;
  const signed char* wvp = wq3 + 2 * (size_t)DIM * DIM;

  v8i accq{}, acck{}, accv{};
  const int trow = tt * 16 + ln;
  const int ocol = ob + ln;

  #pragma unroll
  for (int kt = 0; kt < 8; ++kt) {
    const signed char* ap = xq + (size_t)trow * DIM + kt * 64 + lh * 32;
    const uint2 u0 = *(const uint2*)(ap +  0);
    const uint2 u1 = *(const uint2*)(ap +  8);
    const uint2 u2 = *(const uint2*)(ap + 16);
    const uint2 u3 = *(const uint2*)(ap + 24);
    const v8i a = mk8i2(u0, u1, u2, u3);
    const size_t bo = (size_t)ocol * DIM + kt * 64 + lh * 32;
    const v8i bq = mk8i(*(const uint4*)(wqp + bo), *(const uint4*)(wqp + bo + 16));
    const v8i bk = mk8i(*(const uint4*)(wkp + bo), *(const uint4*)(wkp + bo + 16));
    const v8i bv = mk8i(*(const uint4*)(wvp + bo), *(const uint4*)(wvp + bo + 16));
    accq = __builtin_amdgcn_wmma_i32_16x16x64_iu8(true, a, true, bq, accq, false, false);
    acck = __builtin_amdgcn_wmma_i32_16x16x64_iu8(true, a, true, bk, acck, false, false);
    accv = __builtin_amdgcn_wmma_i32_16x16x64_iu8(true, a, true, bv, accv, false, false);
  }

  const float wsq = wsinv[0], wsk = wsinv[1], wsv = wsinv[2];
  h8 vp;
  #pragma unroll
  for (int r = 0; r < 8; ++r) {
    const int t = tt * 16 + r + lh * 8;
    const float as = ascale[t];
    qf[(size_t)t * DIM + ocol] = (_Float16)((float)accq[r] * as * wsq);
    kf[(size_t)t * DIM + ocol] = (_Float16)((float)acck[r] * as * wsk);
    vp[r] = (_Float16)((float)accv[r] * as * wsv);
  }
  *reinterpret_cast<h8*>(vT + (size_t)ocol * NTOK + tt * 16 + lh * 8) = vp;
}

__global__ __launch_bounds__(128) void k_attn(const _Float16* __restrict__ qf,
                                              const _Float16* __restrict__ kf,
                                              const _Float16* __restrict__ vT,
                                              float* __restrict__ out) {
  const int qt   = blockIdx.x;
  const int b    = blockIdx.y;
  const int lane = threadIdx.x & 31;
  const int w    = threadIdx.x >> 5;
  const int ln   = lane & 15;
  const int lh   = lane >> 4;
  const int db   = w * 128;
  const int q0   = qt * 16;
  const size_t tq0 = (size_t)b * SEQ + q0;

  __shared__ float    sS[2][16 * 32];
  __shared__ _Float16 sP[16 * 32];

  for (int i = threadIdx.x; i < 2 * 16 * 32; i += 128)
    (&sS[0][0])[i] = 0.0f;

  v16h qa[4];
  {
    const _Float16* qp = qf + (tq0 + ln) * DIM + db + lh * 8;
    #pragma unroll
    for (int c = 0; c < 4; ++c)
      qa[c] = mk16h(*(const h8*)(qp + c * 32), *(const h8*)(qp + c * 32 + 16));
  }

  v8f oacc[8] = {};
  float m_run = -1e30f, l_run = 0.0f;

  __syncthreads();

  const int nkb = (q0 + 15) / 32 + 1;
  for (int kb = 0; kb < nkb; ++kb) {
    const int k0 = kb * 32;
    const int p  = kb & 1;

    v16h bv[8];
    #pragma unroll
    for (int nt = 0; nt < 8; ++nt) {
      const _Float16* vp = vT + (size_t)(db + nt * 16 + ln) * NTOK
                         + (size_t)b * SEQ + k0 + lh * 8;
      bv[nt] = mk16h(*(const h8*)(vp), *(const h8*)(vp + 16));
    }

    if (kb + 1 < nkb) {
      const _Float16* pf = kf + ((size_t)b * SEQ + k0 + 32 + lane) * DIM + db;
      __builtin_prefetch(pf, 0, 0);
      __builtin_prefetch(pf + 64, 0, 0);
    }

    #pragma unroll
    for (int nt = 0; nt < 2; ++nt) {
      v8f sacc{};
      const size_t ktok = (size_t)b * SEQ + k0 + nt * 16 + ln;
      const _Float16* kp = kf + ktok * DIM + db + lh * 8;
      #pragma unroll
      for (int c = 0; c < 4; ++c) {
        const v16h bb = mk16h(*(const h8*)(kp + c * 32),
                              *(const h8*)(kp + c * 32 + 16));
        sacc = __builtin_amdgcn_wmma_f32_16x16x32_f16(false, qa[c], false, bb,
                                                      (short)0, sacc, false, false);
      }
      #pragma unroll
      for (int r = 0; r < 8; ++r)
        atomicAdd(&sS[p][(r + lh * 8) * 32 + nt * 16 + ln], sacc[r]);
    }
    __syncthreads();

    float alpha_reg;
    {
      const int rr   = ln;
      const int qrow = q0 + rr;
      const int jb   = lh * 16;
      float sv[16];
      float mloc = -1e30f;
      #pragma unroll
      for (int j = 0; j < 16; ++j) {
        float s = sS[p][rr * 32 + jb + j] * SCORE_SCALE;
        if (k0 + jb + j > qrow) s = -1e30f;
        sv[j] = s;
        mloc = fmaxf(mloc, s);
      }
      const float mnew = fmaxf(m_run, fmaxf(mloc, __shfl_xor(mloc, 16, 32)));
      alpha_reg = __expf(m_run - mnew);
      float lloc = 0.0f;
      #pragma unroll
      for (int j = 0; j < 16; ++j) {
        const float pe = __expf(sv[j] - mnew);
        if (w == 0) sP[rr * 32 + jb + j] = (_Float16)pe;
        lloc += pe;
      }
      lloc += __shfl_xor(lloc, 16, 32);
      l_run = l_run * alpha_reg + lloc;
      m_run = mnew;
      if (w == 0) {
        #pragma unroll
        for (int j = 0; j < 16; ++j) sS[1 - p][rr * 32 + jb + j] = 0.0f;
      }
    }
    __syncthreads();

    #pragma unroll
    for (int r = 0; r < 8; ++r) {
      const float al = __shfl(alpha_reg, r + lh * 8, 32);
      #pragma unroll
      for (int nt = 0; nt < 8; ++nt) oacc[nt][r] *= al;
    }
    v16h pa;
    {
      const _Float16* pp = &sP[ln * 32 + lh * 8];
      pa = mk16h(*(const h8*)(pp), *(const h8*)(pp + 16));
    }
    #pragma unroll
    for (int nt = 0; nt < 8; ++nt)
      oacc[nt] = __builtin_amdgcn_wmma_f32_16x16x32_f16(false, pa, false, bv[nt],
                                                        (short)0, oacc[nt], false, false);
  }

  const float linv = 1.0f / l_run;
  #pragma unroll
  for (int r = 0; r < 8; ++r) {
    const float li = __shfl(linv, r + lh * 8, 32);
    const size_t row = tq0 + r + lh * 8;
    #pragma unroll
    for (int nt = 0; nt < 8; ++nt)
      out[row * DIM + db + nt * 16 + ln] = oacc[nt][r] * li;
  }
}

extern "C" void kernel_launch(void* const* d_in, const int* in_sizes, int n_in,
                              void* d_out, int out_size, void* d_ws, size_t ws_size,
                              hipStream_t stream) {
  (void)in_sizes; (void)n_in; (void)out_size; (void)ws_size;
  const float* x  = (const float*)d_in[0];
  const float* Wq = (const float*)d_in[1];
  const float* Wk = (const float*)d_in[2];
  const float* Wv = (const float*)d_in[3];
  float* out = (float*)d_out;

  char* ws = (char*)d_ws;
  size_t off = 0;
  auto take = [&](size_t bytes) -> char* {
    char* p = ws + off;
    off += (bytes + 255) & ~(size_t)255;
    return p;
  };
  signed char* xq    = (signed char*)take((size_t)NTOK * DIM);
  float*       ascl  = (float*)take((size_t)NTOK * 4);
  signed char* wq3   = (signed char*)take(3ull * DIM * DIM);
  float*       wsinv = (float*)take(256);
  _Float16*    qf    = (_Float16*)take((size_t)NTOK * DIM * 2);
  _Float16*    kf    = (_Float16*)take((size_t)NTOK * DIM * 2);
  _Float16*    vT    = (_Float16*)take((size_t)NTOK * DIM * 2);

  k_actquant<<<NTOK, 128, 0, stream>>>(x, xq, ascl);
  k_wquant<<<3, 256, 0, stream>>>(Wq, Wk, Wv, wq3, wsinv);
  k_proj<<<dim3(NTOK / 16, 4), 256, 0, stream>>>(xq, ascl, wq3, wsinv, qf, kf, vT);
  k_attn<<<dim3(SEQ / 16, NBAT), 128, 0, stream>>>(qf, kf, vT, out);
}
